// MultiHeadedAttention_76003741270281
// MI455X (gfx1250) — matrix operand order repaired and checked by reading, not compiled, not run on hardware
//
#include <hip/hip_runtime.h>
#include <hip/hip_bf16.h>

// ---------------------------------------------------------------------------
// MultiHeadedAttention (strided-window MHA + 3x3 conv + BN + LeakyReLU)
// CDNA5 / gfx1250: bf16 WMMA (v_wmma_f32_16x16x32_bf16) for both GEMM-heavy
// stages; A tiles staged in LDS (ds_read_b128 feeds the matrix pipes);
// weights pre-swizzled into exact B-fragment order (coalesced 32B/lane).
// ---------------------------------------------------------------------------

#define BATCH 8
#define CH    256
#define HH    128
#define WW    128
#define NPIX  (BATCH * HH * WW)       // 131072 pixels
#define LRELU 0.2f
#define BNEPS 1e-5f
#define SLOTP 264                      // per-pixel LDS slot stride (bf16), bank-spread

typedef __bf16 v16bf __attribute__((ext_vector_type(16)));
typedef float  v8f   __attribute__((ext_vector_type(8)));
typedef unsigned int u32x4 __attribute__((ext_vector_type(4)));

union Frag { v16bf v; u32x4 u[2]; };

// ---------------------------------------------------------------------------
// 1) NCHW f32 -> NHWC bf16 transpose (LDS tiled, coalesced both sides)
// grid (B*H, 4), block 256
// ---------------------------------------------------------------------------
__global__ __launch_bounds__(256) void k_pack_x(const float* __restrict__ x,
                                                __bf16* __restrict__ xn) {
  __shared__ float t[64][129];
  const int bh = blockIdx.x;                 // b*128 + h
  const int b = bh >> 7, h = bh & 127;
  const int c0 = blockIdx.y * 64;
#pragma unroll
  for (int i = 0; i < 32; ++i) {             // load: 64c x 128w, coalesced in w
    int idx = threadIdx.x + i * 256;
    int c = idx >> 7, w = idx & 127;
    t[c][w] = x[(((size_t)(b * CH + c0 + c)) * HH + h) * WW + w];
  }
  __syncthreads();
#pragma unroll
  for (int i = 0; i < 32; ++i) {             // store: coalesced in c
    int idx = threadIdx.x + i * 256;
    int w = idx >> 6, c = idx & 63;
    xn[(((size_t)(b * HH + h)) * WW + w) * CH + c0 + c] = (__bf16)t[c][w];
  }
}

// ---------------------------------------------------------------------------
// 2) Pack wq/wk/wv [O][C] f32 -> B-fragment-ordered bf16:
//    [mat3][k8][nt16][lane32][16]  (lane: n = lane&15, K = (e>>3)*16 + (lane>>4)*8 + (e&7))
// ---------------------------------------------------------------------------
__global__ __launch_bounds__(256) void k_pack_wqkv(const float* __restrict__ wq,
                                                   const float* __restrict__ wk,
                                                   const float* __restrict__ wv,
                                                   __bf16* __restrict__ pw) {
  int gid = blockIdx.x * 256 + threadIdx.x;  // 3*8*16*32*16 = 196608
  if (gid >= 196608) return;
  int e = gid & 15, lane = (gid >> 4) & 31, nt = (gid >> 9) & 15;
  int k = (gid >> 13) & 7, mat = gid >> 16;
  int o = nt * 16 + (lane & 15);
  int c = k * 32 + ((e >> 3) << 4) + ((lane >> 4) << 3) + (e & 7);
  const float* src = (mat == 0) ? wq : (mat == 1) ? wk : wv;
  pw[gid] = (__bf16)src[o * CH + c];
}

// ---------------------------------------------------------------------------
// 3) Pack wo [O][C][3][3] -> [tap9][k8][nt16][lane32][16] bf16
// ---------------------------------------------------------------------------
__global__ __launch_bounds__(256) void k_pack_wo(const float* __restrict__ wo,
                                                 __bf16* __restrict__ pwo) {
  int gid = blockIdx.x * 256 + threadIdx.x;  // 9*8*16*32*16 = 589824
  if (gid >= 589824) return;
  int e = gid & 15, lane = (gid >> 4) & 31, nt = (gid >> 9) & 15;
  int k = (gid >> 13) & 7, tap = gid >> 16;
  int o = nt * 16 + (lane & 15);
  int c = k * 32 + ((e >> 3) << 4) + ((lane >> 4) << 3) + (e & 7);
  pwo[gid] = (__bf16)wo[((size_t)o * CH + c) * 9 + tap];
}

// ---------------------------------------------------------------------------
// 4) Fused QKV 1x1-conv GEMM: [NPIX,256] x 3x[256,256]^T, bf16 WMMA.
// grid 8192 (16-pixel M tiles), block 256 = 8 waves; wave owns 6 N-tiles.
// A tile (16pix x 256ch) staged once in LDS; fragments via ds_read_b128.
// ---------------------------------------------------------------------------
__global__ __launch_bounds__(256) void k_qkv(const __bf16* __restrict__ xn,
                                             const __bf16* __restrict__ pw,
                                             const float* __restrict__ bq,
                                             const float* __restrict__ bk,
                                             const float* __restrict__ bv,
                                             __bf16* __restrict__ qn,
                                             __bf16* __restrict__ kn,
                                             __bf16* __restrict__ vn) {
  __shared__ __attribute__((aligned(16))) __bf16 ax[16 * SLOTP];
  const int tid = threadIdx.x;
  const int lane = tid & 31, wid = tid >> 5;
  const int half = lane >> 4, m = lane & 15;
  const size_t p0 = (size_t)blockIdx.x * 16;

  // cooperative stage of A tile: 16 pixels x 256 ch = 512 x 16B chunks
#pragma unroll
  for (int i = 0; i < 2; ++i) {
    int q = tid + i * 256;                   // 0..511
    int pix = q >> 5, ch0 = (q & 31) * 8;
    *(u32x4*)(ax + pix * SLOTP + ch0) =
        *(const u32x4*)(xn + (p0 + pix) * CH + ch0);
  }

  int matj[6], ntj[6], ocj[6];
  v8f acc[6];
#pragma unroll
  for (int j = 0; j < 6; ++j) {
    int nj = wid * 6 + j;                    // 0..47
    matj[j] = nj >> 4;
    ntj[j] = nj & 15;
    ocj[j] = ntj[j] * 16 + m;
    const float* bias = (matj[j] == 0) ? bq : (matj[j] == 1) ? bk : bv;
    float bb = bias[ocj[j]];
#pragma unroll
    for (int r = 0; r < 8; ++r) acc[j][r] = bb;
  }
  __syncthreads();

#pragma unroll
  for (int k = 0; k < 8; ++k) {
    Frag a;
    const __bf16* ap = ax + m * SLOTP + k * 32 + half * 8;
    a.u[0] = *(const u32x4*)ap;
    a.u[1] = *(const u32x4*)(ap + 16);
#pragma unroll
    for (int j = 0; j < 6; ++j) {
      Frag bf;
      const __bf16* bp =
          pw + ((((size_t)matj[j] * 8 + k) * 16 + ntj[j]) * 32 + lane) * 16;
      bf.u[0] = *(const u32x4*)bp;
      bf.u[1] = *(const u32x4*)(bp + 8);
      acc[j] = __builtin_amdgcn_wmma_f32_16x16x32_bf16(
          false, a.v, false, bf.v, (short)0, acc[j], false, false);
    }
  }

#pragma unroll
  for (int j = 0; j < 6; ++j) {
    __bf16* dst = (matj[j] == 0) ? qn : (matj[j] == 1) ? kn : vn;
#pragma unroll
    for (int r = 0; r < 8; ++r) {
      int mm = r + half * 8;                 // D layout: (M = r + half*8, N = lane&15)
      dst[(p0 + mm) * CH + ocj[j]] = (__bf16)acc[j][r];
    }
  }
}

// ---------------------------------------------------------------------------
// 5) Attention scores: one block per (b,s,t), tree-reduce dot over D.
// ---------------------------------------------------------------------------
__global__ __launch_bounds__(256) void k_scores(const __bf16* __restrict__ qn,
                                                const __bf16* __restrict__ kn,
                                                float* __restrict__ sc,
                                                int lo, int lsh, int low,
                                                int D, int S, float scale) {
  const int b = blockIdx.x, s = blockIdx.y, t = blockIdx.z;
  const int sm = (1 << lsh) - 1, owm = (1 << low) - 1;
  const int is = s >> lsh, js = s & sm, it = t >> lsh, jt = t & sm;
  float part = 0.f;
  for (int u = threadIdx.x; u < D; u += 256) {
    int c = u & 63;
    int r = u >> 6;
    int x = r & owm, y = r >> low;
    size_t pq = ((size_t)((b << 7) + (y << lsh) + is) * WW + ((x << lsh) + js)) * CH + lo + c;
    size_t pk = ((size_t)((b << 7) + (y << lsh) + it) * WW + ((x << lsh) + jt)) * CH + lo + c;
    part += (float)qn[pq] * (float)kn[pk];
  }
  __shared__ float red[256];
  red[threadIdx.x] = part;
  __syncthreads();
  for (int o = 128; o > 0; o >>= 1) {
    if ((int)threadIdx.x < o) red[threadIdx.x] += red[threadIdx.x + o];
    __syncthreads();
  }
  if (threadIdx.x == 0) sc[((size_t)b * S + s) * S + t] = red[0] * scale;
}

// ---------------------------------------------------------------------------
// 6) Softmax over rows of S (<=256), one block per (b,s).
// ---------------------------------------------------------------------------
__global__ __launch_bounds__(256) void k_softmax(float* __restrict__ sc, int S) {
  const int b = blockIdx.x, s = blockIdx.y, tid = threadIdx.x;
  float* row = sc + ((size_t)b * S + s) * S;
  __shared__ float red[256];
  float v = (tid < S) ? row[tid] : -3.4e38f;
  red[tid] = v;
  __syncthreads();
  for (int o = 128; o > 0; o >>= 1) {
    if (tid < o) red[tid] = fmaxf(red[tid], red[tid + o]);
    __syncthreads();
  }
  float mx = red[0];
  __syncthreads();
  float e = (tid < S) ? __expf(v - mx) : 0.f;
  red[tid] = e;
  __syncthreads();
  for (int o = 128; o > 0; o >>= 1) {
    if (tid < o) red[tid] += red[tid + o];
    __syncthreads();
  }
  float inv = 1.f / red[0];
  if (tid < S) row[tid] = e * inv;
}

// ---------------------------------------------------------------------------
// 7) y = P x V, V tile (S x 256) staged in LDS; scatter into NHWC bf16 concat
// buffer with the reference's h = i*oh + y, w = j*ow + x layout.
// ---------------------------------------------------------------------------
__global__ __launch_bounds__(256) void k_pv(const __bf16* __restrict__ vn,
                                            const float* __restrict__ p,
                                            __bf16* __restrict__ an,
                                            int lo, int lsh, int low, int S) {
  extern __shared__ __bf16 vt[];
  const int b = blockIdx.x;
  const int d = blockIdx.y * 256 + threadIdx.x;
  const int sm = (1 << lsh) - 1, owm = (1 << low) - 1;
  const int c = d & 63;
  const int r = d >> 6;
  const int x = r & owm, y = r >> low;
  const int oh = HH >> lsh, ow = WW >> lsh;
  for (int t = 0; t < S; ++t) {
    int it = t >> lsh, jt = t & sm;
    vt[t * 256 + threadIdx.x] =
        vn[((size_t)((b << 7) + (y << lsh) + it) * WW + ((x << lsh) + jt)) * CH + lo + c];
  }
  __syncthreads();
  for (int s = 0; s < S; ++s) {
    const float* pr = p + ((size_t)b * S + s) * S;
    float acc = 0.f;
    for (int t = 0; t < S; ++t) acc += pr[t] * (float)vt[t * 256 + threadIdx.x];
    int is = s >> lsh, js = s & sm;
    int hout = is * oh + y, wout = js * ow + x;
    an[((size_t)((b << 7) + hout) * WW + wout) * CH + lo + c] = (__bf16)acc;
  }
}

// ---------------------------------------------------------------------------
// 8) 3x3 conv (SAME, zero pad) as bf16 WMMA GEMM: K = 9 taps x 256 ch.
// grid 8192 (16-pixel tiles), block 256 = 8 waves, wave owns 2 N-tiles.
// Halo tile (3 rows x 18 cols x 256 ch) staged zero-padded in LDS once;
// inner loop is pure ds_read_b128 -> v_wmma with EXEC all-ones.
// ---------------------------------------------------------------------------
__global__ __launch_bounds__(256) void k_conv(const __bf16* __restrict__ an,
                                              const __bf16* __restrict__ pwo,
                                              const float* __restrict__ bo,
                                              float* __restrict__ co) {
  __shared__ __attribute__((aligned(16))) __bf16 at[54 * SLOTP];
  const int tid = threadIdx.x;
  const int lane = tid & 31, wid = tid >> 5;
  const int half = lane >> 4, m = lane & 15;
  const size_t p0 = (size_t)blockIdx.x * 16;
  const int b = (int)(p0 >> 14);
  const int rem = (int)(p0 & 16383);
  const int h = rem >> 7, w0 = rem & 127;

  // zero-padded cooperative stage: 54 slots (3 rows x 18 cols) x 256 ch
#pragma unroll
  for (int i = 0; i < 7; ++i) {
    int q = tid + i * 256;                   // 16B chunks; 1728 total
    if (q < 1728) {
      int slot = q >> 5, ch0 = (q & 31) * 8;
      int rr = slot / 18, cc = slot % 18;
      int hh = h + rr - 1, ww = w0 + cc - 1;
      u32x4 val = (u32x4)0;
      if (hh >= 0 && hh < HH && ww >= 0 && ww < WW)
        val = *(const u32x4*)(an + ((size_t)((b << 7) + hh) * WW + ww) * CH + ch0);
      *(u32x4*)(at + slot * SLOTP + ch0) = val;
    }
  }

  v8f acc[2];
  int ntj[2], ocj[2];
#pragma unroll
  for (int j = 0; j < 2; ++j) {
    ntj[j] = wid * 2 + j;
    ocj[j] = ntj[j] * 16 + m;
    float bb = bo[ocj[j]];
#pragma unroll
    for (int r = 0; r < 8; ++r) acc[j][r] = bb;
  }
  __syncthreads();

#pragma unroll
  for (int tap = 0; tap < 9; ++tap) {
    const int rr = tap / 3, dcol = tap % 3;
    const int slotbase = (rr * 18 + m + dcol) * SLOTP;
    if (tap < 8) {
      // keep next tap's packed-weight block flowing into L2/WGP$
      __builtin_prefetch(
          pwo + ((((size_t)(tap + 1) * 8) * 16 + ntj[0]) * 32 + lane) * 16, 0, 1);
    }
#pragma unroll
    for (int k = 0; k < 8; ++k) {
      Frag a;
      const __bf16* ap = at + slotbase + k * 32 + half * 8;
      a.u[0] = *(const u32x4*)ap;
      a.u[1] = *(const u32x4*)(ap + 16);
#pragma unroll
      for (int j = 0; j < 2; ++j) {
        Frag bf;
        const __bf16* bp =
            pwo + ((((size_t)tap * 8 + k) * 16 + ntj[j]) * 32 + lane) * 16;
        bf.u[0] = *(const u32x4*)bp;
        bf.u[1] = *(const u32x4*)(bp + 8);
        acc[j] = __builtin_amdgcn_wmma_f32_16x16x32_bf16(
            false, a.v, false, bf.v, (short)0, acc[j], false, false);
      }
    }
  }
#pragma unroll
  for (int j = 0; j < 2; ++j)
#pragma unroll
    for (int r = 0; r < 8; ++r) {
      int mm = r + half * 8;
      co[(p0 + mm) * CH + ocj[j]] = acc[j][r];
    }
}

// ---------------------------------------------------------------------------
// 9) BN stats, deterministic two-pass (no float atomics).
// ---------------------------------------------------------------------------
__global__ __launch_bounds__(256) void k_bnstats1(const float* __restrict__ co,
                                                  float* __restrict__ psum,
                                                  float* __restrict__ psq) {
  int g = blockIdx.x * 256 + threadIdx.x;    // 131072 threads
  int c = g & 255;
  int slot = g >> 8;                         // 0..511
  float s = 0.f, s2 = 0.f;
  size_t pbase = (size_t)slot * 256;         // 256 pixels per slot
  for (int i = 0; i < 256; ++i) {
    float v = co[(pbase + i) * CH + c];
    s += v;
    s2 += v * v;
  }
  psum[(size_t)slot * CH + c] = s;
  psq[(size_t)slot * CH + c] = s2;
}

__global__ void k_bnstats2(const float* __restrict__ psum,
                           const float* __restrict__ psq,
                           float* __restrict__ stats) {
  int c = threadIdx.x;                       // 256 threads, 1 block
  float s = 0.f, s2 = 0.f;
  for (int slot = 0; slot < 512; ++slot) {
    s += psum[(size_t)slot * CH + c];
    s2 += psq[(size_t)slot * CH + c];
  }
  stats[c] = s;
  stats[CH + c] = s2;
}

// ---------------------------------------------------------------------------
// 10) BN apply + LeakyReLU + NHWC f32 -> NCHW f32 output transpose.
// ---------------------------------------------------------------------------
__global__ __launch_bounds__(256) void k_final(const float* __restrict__ co,
                                               const float* __restrict__ stats,
                                               const float* __restrict__ gamma,
                                               const float* __restrict__ beta,
                                               float* __restrict__ out) {
  __shared__ float tl[64][129];
  const int bh = blockIdx.x, b = bh >> 7, h = bh & 127;
  const int c0 = blockIdx.y * 64;
#pragma unroll
  for (int i = 0; i < 32; ++i) {             // coalesced NHWC load
    int idx = threadIdx.x + i * 256;
    int w = idx >> 6, c = idx & 63;
    tl[c][w] = co[((size_t)((b << 7) + h) * WW + w) * CH + c0 + c];
  }
  __syncthreads();
  const float invN = 1.f / (float)NPIX;
#pragma unroll
  for (int i = 0; i < 32; ++i) {             // coalesced NCHW store
    int idx = threadIdx.x + i * 256;
    int c = idx >> 7, w = idx & 127;
    int ch = c0 + c;
    float mean = stats[ch] * invN;
    float var = stats[CH + ch] * invN - mean * mean;
    float inv = rsqrtf(var + BNEPS);
    float g = gamma[ch] * inv;
    float bb = beta[ch] - mean * g;
    float val = tl[c][w] * g + bb;
    out[((size_t)(b * CH + ch) * HH + h) * WW + w] = (val > 0.f) ? val : LRELU * val;
  }
}

// ---------------------------------------------------------------------------
// Host: workspace carve + launch sequence (all on `stream`, graph-safe).
// ---------------------------------------------------------------------------
extern "C" void kernel_launch(void* const* d_in, const int* in_sizes, int n_in,
                              void* d_out, int out_size, void* d_ws, size_t ws_size,
                              hipStream_t stream) {
  const float* x     = (const float*)d_in[0];
  const float* wq    = (const float*)d_in[1];
  const float* bq    = (const float*)d_in[2];
  const float* wk    = (const float*)d_in[3];
  const float* bk    = (const float*)d_in[4];
  const float* wv    = (const float*)d_in[5];
  const float* bv    = (const float*)d_in[6];
  const float* wo    = (const float*)d_in[7];
  const float* bo    = (const float*)d_in[8];
  const float* gamma = (const float*)d_in[9];
  const float* beta  = (const float*)d_in[10];
  float* out = (float*)d_out;

  // workspace carve (256B aligned)
  char* wp = (char*)d_ws;
  auto carve = [&](size_t bytes) -> void* {
    void* r = (void*)wp;
    wp += (bytes + 255) & ~(size_t)255;
    return r;
  };
  const size_t nhwc = (size_t)NPIX * CH;
  __bf16* xn  = (__bf16*)carve(nhwc * 2);           // x, NHWC bf16
  __bf16* qn  = (__bf16*)carve(nhwc * 2);
  __bf16* kn  = (__bf16*)carve(nhwc * 2);
  __bf16* vn  = (__bf16*)carve(nhwc * 2);
  __bf16* an  = (__bf16*)carve(nhwc * 2);           // attention concat, NHWC bf16
  float*  co  = (float*)carve(nhwc * 4);            // conv out, NHWC f32
  __bf16* pw  = (__bf16*)carve((size_t)196608 * 2); // packed wq/wk/wv frags
  __bf16* pwo = (__bf16*)carve((size_t)589824 * 2); // packed wo frags
  float*  sc  = (float*)carve((size_t)8 * 69904 * 4); // scores/probs (all strides)
  float*  psum = (float*)carve((size_t)512 * CH * 4);
  float*  psq  = (float*)carve((size_t)512 * CH * 4);
  float*  stats = (float*)carve(2 * CH * 4);

  // 1-3: layout conversions / weight packing
  k_pack_x<<<dim3(BATCH * HH, 4), 256, 0, stream>>>(x, xn);
  k_pack_wqkv<<<768, 256, 0, stream>>>(wq, wk, wv, pw);
  k_pack_wo<<<2304, 256, 0, stream>>>(wo, pwo);

  // 4: fused QKV GEMM (WMMA)
  k_qkv<<<NPIX / 16, 256, 0, stream>>>(xn, pw, bq, bk, bv, qn, kn, vn);

  // 5-7: per-stride attention
  size_t scoff = 0;
  for (int idx = 0; idx < 4; ++idx) {
    const int lsh = idx + 1;                 // stride = 1<<lsh
    const int low = 7 - lsh;                 // log2(ow)
    const int S = 1 << (2 * lsh);            // stride^2
    const int D = 64 * (HH >> lsh) * (WW >> lsh);
    const int lo = idx * 64;
    const float scale = 1.f / sqrtf((float)D);
    float* scp = sc + scoff;
    scoff += (size_t)BATCH * S * S;
    k_scores<<<dim3(BATCH, S, S), 256, 0, stream>>>(qn, kn, scp, lo, lsh, low, D, S, scale);
    k_softmax<<<dim3(BATCH, S), 256, 0, stream>>>(scp, S);
    k_pv<<<dim3(BATCH, D / 256), 256, (size_t)S * 256 * 2, stream>>>(vn, scp, an, lo, lsh, low, S);
  }

  // 8: 3x3 conv (WMMA)
  k_conv<<<NPIX / 16, 256, 0, stream>>>(an, pwo, bo, co);

  // 9: BN stats (deterministic)
  k_bnstats1<<<512, 256, 0, stream>>>(co, psum, psq);
  k_bnstats2<<<1, 256, 0, stream>>>(psum, psq, stats);

  // 10: BN + LeakyReLU + NCHW output
  k_final<<<dim3(BATCH * HH, 4), 256, 0, stream>>>(co, stats, gamma, beta, out);
}
